// Qwen3Attention_60756607369506
// MI455X (gfx1250) — matrix operand order repaired and checked by reading, not compiled, not run on hardware
//
#include <hip/hip_runtime.h>
#include <cstdint>
#include <cstddef>

#define BB   2
#define SS   2048
#define HH   2048
#define NH   16
#define NKV  8
#define HD   128
#define NREP (NH / NKV)
#define EPSF 1e-6f
#define SCALEF 0.08838834764831845f   // 1/sqrt(128)

typedef __attribute__((ext_vector_type(16))) __bf16 v16bf;
typedef __attribute__((ext_vector_type(8)))  float  v8f;

union Frag16 { v16bf v; uint4 u[2]; };

// ---------------------------------------------------------------------------
// f32 -> bf16 bulk convert (4 elements / thread)
// ---------------------------------------------------------------------------
__global__ __launch_bounds__(256) void cvt_bf16_kernel(const float* __restrict__ src,
                                                       __bf16* __restrict__ dst, int n4) {
    int i = blockIdx.x * 256 + threadIdx.x;
    if (i < n4) {
        float4 f = reinterpret_cast<const float4*>(src)[i];
        __bf16* d = dst + (size_t)i * 4;
        d[0] = (__bf16)f.x; d[1] = (__bf16)f.y; d[2] = (__bf16)f.z; d[3] = (__bf16)f.w;
    }
}

// ---------------------------------------------------------------------------
// C[M,N] = A[M,K] * W[N,K]^T (bf16 in, f32 out).
// Wave macro-tile: 32x64 (2 M-tiles x 4 N-tiles, 8 accumulators).
// Block = 8 waves as 2(M) x 4(N)  ->  block tile 64 x 256.
// grid = (N/256, M/64)
// ---------------------------------------------------------------------------
__global__ __launch_bounds__(256) void gemm_bf16_nt(const __bf16* __restrict__ A,
                                                    const __bf16* __restrict__ Wt,
                                                    float* __restrict__ C,
                                                    int M, int N, int K) {
    const int lane = threadIdx.x & 31;
    const int wid  = threadIdx.x >> 5;
    const int wy   = wid >> 2;                 // 0..1
    const int wx   = wid & 3;                  // 0..3
    const int kh   = lane >> 4;                // K-half select (16-bit frag layout)
    const int l15  = lane & 15;

    const int mbase = blockIdx.y * 4 + wy * 2; // in 16-row tiles
    const int nbase = blockIdx.x * 16 + wx * 4;

    const __bf16* ap[2];
    const __bf16* bp[4];
#pragma unroll
    for (int mt = 0; mt < 2; ++mt)
        ap[mt] = A + (size_t)((mbase + mt) * 16 + l15) * K + kh * 8;
#pragma unroll
    for (int nt = 0; nt < 4; ++nt)
        bp[nt] = Wt + (size_t)((nbase + nt) * 16 + l15) * K + kh * 8;

    v8f acc[2][4];
#pragma unroll
    for (int mt = 0; mt < 2; ++mt)
#pragma unroll
        for (int nt = 0; nt < 4; ++nt) acc[mt][nt] = {};

    for (int k0 = 0; k0 < K; k0 += 32) {
        Frag16 a[2], b[4];
#pragma unroll
        for (int mt = 0; mt < 2; ++mt) {
            a[mt].u[0] = *reinterpret_cast<const uint4*>(ap[mt] + k0);
            a[mt].u[1] = *reinterpret_cast<const uint4*>(ap[mt] + k0 + 16);
        }
#pragma unroll
        for (int nt = 0; nt < 4; ++nt) {
            b[nt].u[0] = *reinterpret_cast<const uint4*>(bp[nt] + k0);
            b[nt].u[1] = *reinterpret_cast<const uint4*>(bp[nt] + k0 + 16);
        }
#pragma unroll
        for (int mt = 0; mt < 2; ++mt)
#pragma unroll
            for (int nt = 0; nt < 4; ++nt)
                acc[mt][nt] = __builtin_amdgcn_wmma_f32_16x16x32_bf16(
                    false, a[mt].v, false, b[nt].v, (short)0, acc[mt][nt], false, false);
    }

#pragma unroll
    for (int mt = 0; mt < 2; ++mt) {
        float* cp = C + (size_t)((mbase + mt) * 16 + kh * 8) * N;
#pragma unroll
        for (int nt = 0; nt < 4; ++nt) {
            const int n = (nbase + nt) * 16 + l15;
#pragma unroll
            for (int j = 0; j < 8; ++j) cp[(size_t)j * N + n] = acc[mt][nt][j];
        }
    }
}

// ---------------------------------------------------------------------------
// Per-head RMSNorm + RoPE; (b,s,h,d) f32 -> (b,h,s,d) bf16. One wave per row.
// ---------------------------------------------------------------------------
__global__ __launch_bounds__(256) void norm_rope_kernel(const float* __restrict__ X,
                                                        const float* __restrict__ cosb,
                                                        const float* __restrict__ sinb,
                                                        const float* __restrict__ nw,
                                                        __bf16* __restrict__ Out,
                                                        int nheads) {
    const int row  = blockIdx.x * 8 + (threadIdx.x >> 5);   // (b*S + s)*nheads + h
    const int lane = threadIdx.x & 31;
    const int h = row % nheads;
    const int s = (row / nheads) % SS;
    const int b = row / (nheads * SS);

    const float* xp = X + (size_t)row * HD + lane * 4;
    float4 x = *reinterpret_cast<const float4*>(xp);

    float ssum = x.x * x.x + x.y * x.y + x.z * x.z + x.w * x.w;
#pragma unroll
    for (int m = 1; m < 32; m <<= 1) ssum += __shfl_xor(ssum, m, 32);
    const float inv = rsqrtf(ssum * (1.0f / HD) + EPSF);

    float4 w = *reinterpret_cast<const float4*>(nw + lane * 4);
    float4 xn = { x.x * inv * w.x, x.y * inv * w.y, x.z * inv * w.z, x.w * inv * w.w };

    const size_t cso = ((size_t)b * SS + s) * HD + lane * 4;
    float4 c  = *reinterpret_cast<const float4*>(cosb + cso);
    float4 sn = *reinterpret_cast<const float4*>(sinb + cso);

    float4 other;                        // partner element 64 apart = lane ^ 16
    other.x = __shfl_xor(xn.x, 16, 32);
    other.y = __shfl_xor(xn.y, 16, 32);
    other.z = __shfl_xor(xn.z, 16, 32);
    other.w = __shfl_xor(xn.w, 16, 32);
    const float sg = (lane < 16) ? -1.0f : 1.0f;   // rot_half = (-x2, x1)

    float4 r = { xn.x * c.x + sg * other.x * sn.x,
                 xn.y * c.y + sg * other.y * sn.y,
                 xn.z * c.z + sg * other.z * sn.z,
                 xn.w * c.w + sg * other.w * sn.w };

    __bf16* op = Out + (((size_t)(b * nheads + h) * SS + s) * HD) + lane * 4;
    op[0] = (__bf16)r.x; op[1] = (__bf16)r.y; op[2] = (__bf16)r.z; op[3] = (__bf16)r.w;
}

// ---------------------------------------------------------------------------
// V: (b,s,kvh,d) f32 -> (b,kvh,d,s) bf16 (transposed for contiguous B-frags)
// ---------------------------------------------------------------------------
__global__ __launch_bounds__(256) void v_transpose_kernel(const float* __restrict__ Vf,
                                                          __bf16* __restrict__ Vt) {
    size_t i = (size_t)blockIdx.x * 256 + threadIdx.x;   // B*S*NKV*HD elements
    int d = (int)(i % HD);
    size_t t = i / HD;
    int kvh = (int)(t % NKV);
    size_t sb = t / NKV;
    int s = (int)(sb % SS);
    int b = (int)(sb / SS);
    Vt[(((size_t)(b * NKV + kvh)) * HD + d) * SS + s] = (__bf16)Vf[i];
}

// ---------------------------------------------------------------------------
// FUSED scores + softmax:
//   one block = 16 queries x ALL 2048 keys for one (b,h).
//   8 waves x 16 accumulators hold the whole 16x2048 strip in VGPRs.
//   Causal: tiles with ntile > mtile are skipped (exact 0 after softmax).
//   Row max/sum: shfl_xor within 16-lane row groups + 512B LDS across waves.
// grid = (S/16, B*NH), block = 256
// ---------------------------------------------------------------------------
__global__ __launch_bounds__(256) void attn_fused_kernel(const __bf16* __restrict__ Q,
                                                         const __bf16* __restrict__ Km,
                                                         const float* __restrict__ mask,
                                                         float* __restrict__ Wts) {
    const int lane = threadIdx.x & 31;
    const int wid  = threadIdx.x >> 5;
    const int kh   = lane >> 4;
    const int l15  = lane & 15;

    const int mtile = blockIdx.x;
    const int bh    = blockIdx.y;
    const int b = bh / NH, h = bh % NH, kvh = h / NREP;

    const __bf16* qp = Q + ((size_t)(b * NH + h) * SS + mtile * 16 + l15) * HD + kh * 8;
    const __bf16* kb = Km + ((size_t)(b * NKV + kvh) * SS + wid * 256 + l15) * HD + kh * 8;

    v8f acc[16];
#pragma unroll
    for (int t = 0; t < 16; ++t) acc[t] = {};

#pragma unroll
    for (int k0 = 0; k0 < HD; k0 += 32) {
        Frag16 a;
        a.u[0] = *reinterpret_cast<const uint4*>(qp + k0);
        a.u[1] = *reinterpret_cast<const uint4*>(qp + k0 + 16);
#pragma unroll
        for (int t = 0; t < 16; ++t) {
            if (wid * 16 + t <= mtile) {               // causal skip (wave-uniform)
                Frag16 bf;
                const __bf16* kp = kb + (size_t)t * (16 * HD) + k0;
                bf.u[0] = *reinterpret_cast<const uint4*>(kp);
                bf.u[1] = *reinterpret_cast<const uint4*>(kp + 16);
                acc[t] = __builtin_amdgcn_wmma_f32_16x16x32_bf16(
                    false, a.v, false, bf.v, (short)0, acc[t], false, false);
            }
        }
    }

    const int mrow = mtile * 16 + kh * 8;   // first of this lane's 8 rows

    // scale + mask, per-lane row max over live tiles
    float pmax[8];
#pragma unroll
    for (int j = 0; j < 8; ++j) pmax[j] = -3.4e38f;
#pragma unroll
    for (int t = 0; t < 16; ++t) {
        if (wid * 16 + t <= mtile) {
            const int n = (wid * 16 + t) * 16 + l15;
#pragma unroll
            for (int j = 0; j < 8; ++j) {
                float v = acc[t][j] * SCALEF + mask[((size_t)b * SS + mrow + j) * SS + n];
                acc[t][j] = v;
                pmax[j] = fmaxf(pmax[j], v);
            }
        }
    }
#pragma unroll
    for (int j = 0; j < 8; ++j) {           // reduce within 16-lane row group
        pmax[j] = fmaxf(pmax[j], __shfl_xor(pmax[j], 1, 32));
        pmax[j] = fmaxf(pmax[j], __shfl_xor(pmax[j], 2, 32));
        pmax[j] = fmaxf(pmax[j], __shfl_xor(pmax[j], 4, 32));
        pmax[j] = fmaxf(pmax[j], __shfl_xor(pmax[j], 8, 32));
    }

    __shared__ float red[16][8];            // [row][wave]
    if (l15 == 0) {
#pragma unroll
        for (int j = 0; j < 8; ++j) red[kh * 8 + j][wid] = pmax[j];
    }
    __syncthreads();
    float gmax[8];
#pragma unroll
    for (int j = 0; j < 8; ++j) {
        float4 r0 = *reinterpret_cast<const float4*>(&red[kh * 8 + j][0]);
        float4 r1 = *reinterpret_cast<const float4*>(&red[kh * 8 + j][4]);
        gmax[j] = fmaxf(fmaxf(fmaxf(r0.x, r0.y), fmaxf(r0.z, r0.w)),
                        fmaxf(fmaxf(r1.x, r1.y), fmaxf(r1.z, r1.w)));
    }
    __syncthreads();                        // before red[] reuse

    float psum[8];
#pragma unroll
    for (int j = 0; j < 8; ++j) psum[j] = 0.0f;
#pragma unroll
    for (int t = 0; t < 16; ++t) {
        if (wid * 16 + t <= mtile) {
#pragma unroll
            for (int j = 0; j < 8; ++j) {
                float e = __expf(acc[t][j] - gmax[j]);
                acc[t][j] = e;
                psum[j] += e;
            }
        }
    }
#pragma unroll
    for (int j = 0; j < 8; ++j) {
        psum[j] += __shfl_xor(psum[j], 1, 32);
        psum[j] += __shfl_xor(psum[j], 2, 32);
        psum[j] += __shfl_xor(psum[j], 4, 32);
        psum[j] += __shfl_xor(psum[j], 8, 32);
    }
    if (l15 == 0) {
#pragma unroll
        for (int j = 0; j < 8; ++j) red[kh * 8 + j][wid] = psum[j];
    }
    __syncthreads();
    float ginv[8];
#pragma unroll
    for (int j = 0; j < 8; ++j) {
        float4 r0 = *reinterpret_cast<const float4*>(&red[kh * 8 + j][0]);
        float4 r1 = *reinterpret_cast<const float4*>(&red[kh * 8 + j][4]);
        ginv[j] = 1.0f / ((r0.x + r0.y + r0.z + r0.w) + (r1.x + r1.y + r1.z + r1.w));
    }

    // single write of the final softmaxed weights (zeros above the diagonal)
#pragma unroll
    for (int t = 0; t < 16; ++t) {
        const int n = (wid * 16 + t) * 16 + l15;
        const bool live = (wid * 16 + t <= mtile);
#pragma unroll
        for (int j = 0; j < 8; ++j)
            Wts[((size_t)bh * SS + mrow + j) * SS + n] = live ? acc[t][j] * ginv[j] : 0.0f;
    }
}

// ---------------------------------------------------------------------------
// Heads = P(f32, per (b,h) SxS) x V^T(b,kvh,HD,S) -> bf16 (b,s,NH*HD).
// Wave tile: 16(M) x 128(full head dim); P-frag converted f32->bf16 once per
// k-step; k-loop stops at the causal boundary (weights beyond are exact 0).
// grid = (S/128, B*NH), block = 256 (8 waves = 8 M-tiles)
// ---------------------------------------------------------------------------
__global__ __launch_bounds__(256) void attn_pv_kernel(const float* __restrict__ Wts,
                                                      const __bf16* __restrict__ Vt,
                                                      __bf16* __restrict__ Heads) {
    const int lane = threadIdx.x & 31;
    const int wid  = threadIdx.x >> 5;
    const int kh   = lane >> 4;
    const int l15  = lane & 15;

    const int mtile = blockIdx.x * 8 + wid;
    const int bh    = blockIdx.y;
    const int b = bh / NH, h = bh % NH, kvh = h / NREP;

    const float* ap = Wts + ((size_t)bh * SS + mtile * 16 + l15) * SS + kh * 8;
    const __bf16* vb = Vt + ((size_t)(b * NKV + kvh) * HD + l15) * SS + kh * 8;

    v8f acc[8];
#pragma unroll
    for (int nt = 0; nt < 8; ++nt) acc[nt] = {};

    const int kend = (mtile * 16 + 16 + 31) & ~31;   // causal: P[m][k>=kend] == 0
    for (int k0 = 0; k0 < kend; k0 += 32) {
        Frag16 a;
        const float* fp = ap + k0;
#pragma unroll
        for (int i = 0; i < 8; ++i) a.v[i]     = (__bf16)fp[i];
#pragma unroll
        for (int i = 0; i < 8; ++i) a.v[8 + i] = (__bf16)fp[16 + i];
#pragma unroll
        for (int nt = 0; nt < 8; ++nt) {
            Frag16 bf;
            const __bf16* bp = vb + (size_t)nt * 16 * SS + k0;   // const stride -> imm offsets
            bf.u[0] = *reinterpret_cast<const uint4*>(bp);
            bf.u[1] = *reinterpret_cast<const uint4*>(bp + 16);
            acc[nt] = __builtin_amdgcn_wmma_f32_16x16x32_bf16(
                false, a.v, false, bf.v, (short)0, acc[nt], false, false);
        }
    }

#pragma unroll
    for (int nt = 0; nt < 8; ++nt) {
        const int n = nt * 16 + l15;
#pragma unroll
        for (int j = 0; j < 8; ++j) {
            const int m = mtile * 16 + kh * 8 + j;
            Heads[((size_t)b * SS + m) * (NH * HD) + h * HD + n] = (__bf16)acc[nt][j];
        }
    }
}

// ---------------------------------------------------------------------------
extern "C" void kernel_launch(void* const* d_in, const int* in_sizes, int n_in,
                              void* d_out, int out_size, void* d_ws, size_t ws_size,
                              hipStream_t stream) {
    const float* hs   = (const float*)d_in[0];
    const float* cosb = (const float*)d_in[1];
    const float* sinb = (const float*)d_in[2];
    const float* mask = (const float*)d_in[3];
    const float* Wq   = (const float*)d_in[4];
    const float* Wk   = (const float*)d_in[5];
    const float* Wv   = (const float*)d_in[6];
    const float* Wo   = (const float*)d_in[7];
    const float* qnw  = (const float*)d_in[8];
    const float* knw  = (const float*)d_in[9];

    char* ws = (char*)d_ws;
    __bf16* hs_bf = (__bf16*)(ws + 0);                    // 16 MB
    __bf16* wq_bf = (__bf16*)(ws + 16777216);             //  8 MB
    __bf16* wk_bf = (__bf16*)(ws + 25165824);             //  4 MB
    __bf16* wv_bf = (__bf16*)(ws + 29360128);             //  4 MB
    __bf16* wo_bf = (__bf16*)(ws + 33554432);             //  8 MB
    float*  qf32  = (float*) (ws + 41943040);             // 32 MB
    float*  kf32  = (float*) (ws + 75497472);             // 16 MB
    float*  vf32  = (float*) (ws + 92274688);             // 16 MB
    __bf16* q_bf  = (__bf16*)(ws + 109051904);            // 16 MB
    __bf16* k_bf  = (__bf16*)(ws + 125829120);            //  8 MB
    __bf16* vT_bf = (__bf16*)(ws + 134217728);            //  8 MB
    __bf16* hd_bf = (__bf16*)(ws + 41943040);             // alias qf32 (dead by then)

    float* attn_out = (float*)d_out;                                // B*S*H
    float* attn_w   = (float*)d_out + (size_t)BB * SS * HH;         // B*NH*S*S

    const int M = BB * SS;   // 4096

    // 1) convert to bf16
    cvt_bf16_kernel<<<(M * HH / 4 + 255) / 256, 256, 0, stream>>>(hs, hs_bf, M * HH / 4);
    cvt_bf16_kernel<<<(HH * HH / 4 + 255) / 256, 256, 0, stream>>>(Wq, wq_bf, HH * HH / 4);
    cvt_bf16_kernel<<<(NKV * HD * HH / 4 + 255) / 256, 256, 0, stream>>>(Wk, wk_bf, NKV * HD * HH / 4);
    cvt_bf16_kernel<<<(NKV * HD * HH / 4 + 255) / 256, 256, 0, stream>>>(Wv, wv_bf, NKV * HD * HH / 4);
    cvt_bf16_kernel<<<(HH * HH / 4 + 255) / 256, 256, 0, stream>>>(Wo, wo_bf, HH * HH / 4);

    // 2) QKV projections (WMMA) — block tile 64x256
    gemm_bf16_nt<<<dim3(NH * HD / 256, M / 64), 256, 0, stream>>>(hs_bf, wq_bf, qf32, M, NH * HD, HH);
    gemm_bf16_nt<<<dim3(NKV * HD / 256, M / 64), 256, 0, stream>>>(hs_bf, wk_bf, kf32, M, NKV * HD, HH);
    gemm_bf16_nt<<<dim3(NKV * HD / 256, M / 64), 256, 0, stream>>>(hs_bf, wv_bf, vf32, M, NKV * HD, HH);

    // 3) RMSNorm + RoPE (Q, K) ; V transpose
    norm_rope_kernel<<<(BB * SS * NH) / 8, 256, 0, stream>>>(qf32, cosb, sinb, qnw, q_bf, NH);
    norm_rope_kernel<<<(BB * SS * NKV) / 8, 256, 0, stream>>>(kf32, cosb, sinb, knw, k_bf, NKV);
    v_transpose_kernel<<<(BB * SS * NKV * HD) / 256, 256, 0, stream>>>(vf32, vT_bf);

    // 4) fused scores + softmax (WMMA) -> final attn_weights in d_out
    attn_fused_kernel<<<dim3(SS / 16, BB * NH), 256, 0, stream>>>(q_bf, k_bf, mask, attn_w);

    // 5) heads = P * V (WMMA, causal-trimmed k-loop)
    attn_pv_kernel<<<dim3(SS / 128, BB * NH), 256, 0, stream>>>(attn_w, vT_bf, hd_bf);

    // 6) output projection (WMMA) -> attn_output
    gemm_bf16_nt<<<dim3(HH / 256, M / 64), 256, 0, stream>>>(hd_bf, wo_bf, attn_out, M, HH, NH * HD);
}
